// DecoderRNN_83958020702364
// MI455X (gfx1250) — matrix operand order repaired and checked by reading, not compiled, not run on hardware
//
#include <hip/hip_runtime.h>
#include <hip/hip_bf16.h>
#include <stdint.h>

// Problem constants (match reference)
#define BB 32
#define TT 32
#define LL 64
#define EMB 256
#define HID 512
#define H4 2048
#define KC 768      // EMB + HID  (LSTM gate GEMM K)
#define VOCAB 30000
#define NOBJ 91
#define KF 576      // HID + LL   (fused output GEMM K)

typedef __attribute__((ext_vector_type(16))) __bf16 v16bf;
typedef __attribute__((ext_vector_type(8)))  float  v8f;
typedef __attribute__((ext_vector_type(4)))  int    v4i;

union Frag16 { v16bf v; uint16_t u[16]; uint4 q[2]; };

// ---- CDNA5 async global->LDS path (guarded; falls back to VGPR staging) ----
#if defined(__AMDGCN__) && __has_builtin(__builtin_amdgcn_global_load_async_to_lds_b128) && \
    __has_builtin(__builtin_amdgcn_s_wait_asynccnt)
#define HAVE_ASYNC_LDS 1
#else
#define HAVE_ASYNC_LDS 0
#endif

#define GLOBAL_AS __attribute__((address_space(1)))
#define LDS_AS    __attribute__((address_space(3)))

__device__ __forceinline__ void async_cp16(const void* g, void* l){
#if HAVE_ASYNC_LDS
  __builtin_amdgcn_global_load_async_to_lds_b128(
      (GLOBAL_AS v4i*)g, (LDS_AS v4i*)l, 0, 0);
#else
  *(uint4*)l = *(const uint4*)g;
#endif
}
__device__ __forceinline__ void wait_async(){
#if HAVE_ASYNC_LDS
  __builtin_amdgcn_s_wait_asynccnt(0);
#endif
}

__device__ __forceinline__ uint16_t f2bf(float f) {
  uint32_t u = __float_as_uint(f);
  u += 0x7FFFu + ((u >> 16) & 1u);      // round-to-nearest-even
  return (uint16_t)(u >> 16);
}
__device__ __forceinline__ float sigmoidf_(float x){ return 1.f/(1.f+__expf(-x)); }

// ---------------- prep kernels ----------------

__global__ void k_cvt_bf16(const float* __restrict__ src, uint16_t* __restrict__ dst, int n){
  int i = blockIdx.x*blockDim.x + threadIdx.x;
  int stride = gridDim.x*blockDim.x;
  for(; i<n; i+=stride) dst[i] = f2bf(src[i]);
}

// WgT[n][k]: n in [0,2048) output gate channel, k in [0,768): [W_ih row | W_hh row]
__global__ void k_prep_wgt(const float* __restrict__ Wih, const float* __restrict__ Whh,
                           uint16_t* __restrict__ wgt){
  int i = blockIdx.x*blockDim.x + threadIdx.x;
  int stride = gridDim.x*blockDim.x;
  for(; i<H4*KC; i+=stride){
    int n = i / KC, k = i % KC;
    float v = (k < EMB) ? Wih[(size_t)n*EMB + k] : Whh[(size_t)n*HID + (k-EMB)];
    wgt[i] = f2bf(v);
  }
}

// embB[t][b][e]: t=0 -> features, t>=1 -> embed_W[captions[b][t-1]]
__global__ void k_prep_emb(const float* __restrict__ feat, const int* __restrict__ caps,
                           const float* __restrict__ embW, uint16_t* __restrict__ embB){
  int i = blockIdx.x*blockDim.x + threadIdx.x;
  int stride = gridDim.x*blockDim.x;
  for(; i<TT*BB*EMB; i+=stride){
    int t = i/(BB*EMB); int r = i%(BB*EMB); int b = r/EMB; int e = r%EMB;
    float v = (t==0) ? feat[(size_t)b*EMB+e]
                     : embW[(size_t)caps[b*TT + (t-1)]*EMB + e];
    embB[i] = f2bf(v);
  }
}

__global__ void k_init_state(uint16_t* __restrict__ hB, float* __restrict__ cF){
  int i = blockIdx.x*blockDim.x + threadIdx.x;
  int stride = gridDim.x*blockDim.x;
  for(; i<BB*HID; i+=stride){ hB[i] = 0; cF[i] = 0.f; }
}

// ---------------- LSTM: gate GEMM via WMMA bf16 ----------------
// C[32 x 2048] = Xcat[32 x 768] @ WgT^T ; block tile = 32(M) x 64(N), 4 waves
__global__ __launch_bounds__(128) void k_lstm_gemm(const uint16_t* __restrict__ embB,
                                                   const uint16_t* __restrict__ hB,
                                                   const uint16_t* __restrict__ wgt,
                                                   float* __restrict__ gates, int t){
  __shared__ uint16_t sB[64*32];           // [n][k] tile, 4KB
  const int tid  = threadIdx.x;
  const int lane = tid & 31;
  const int wave = tid >> 5;
  const int n0b  = blockIdx.x * 64;
  const int n0w  = wave * 16;
  v8f c0 = {}; v8f c1 = {};
  for(int kk=0; kk<KC/32; ++kk){
    const int k0 = kk*32;
    { // stage B panel: async 16B per thread, [N][K] global -> [n][k] LDS
      int n = tid >> 1, half = tid & 1;
      async_cp16(wgt + (size_t)(n0b+n)*KC + k0 + half*16, &sB[n*32 + half*16]);
    }
    wait_async();
    __syncthreads();
    Frag16 bf, a0, a1;
    { // B frag: lane=column n; K 8-granular interleave (0..7 / +16..23)
      int n   = n0w + (lane & 15);
      int kbB = (lane >> 4) * 8;
      bf.q[0] = *(const uint4*)&sB[n*32 + kbB];
      bf.q[1] = *(const uint4*)&sB[n*32 + kbB + 16];
    }
    { // A frag: lane=row m; K 8-granular interleave (0..7 / +16..23)
      int kbA = (lane >> 4) * 8;
      int r0  = lane & 15;
      const uint16_t *p0, *p1;
      if (k0 < EMB){
        p0 = embB + ((size_t)(t*BB + r0     )*EMB + k0 + kbA);
        p1 = embB + ((size_t)(t*BB + r0 + 16)*EMB + k0 + kbA);
      } else {
        p0 = hB + ((size_t)(r0     )*HID + (k0-EMB) + kbA);
        p1 = hB + ((size_t)(r0 + 16)*HID + (k0-EMB) + kbA);
      }
      a0.q[0] = *(const uint4*)p0; a0.q[1] = *(const uint4*)(p0+16);
      a1.q[0] = *(const uint4*)p1; a1.q[1] = *(const uint4*)(p1+16);
    }
    c0 = __builtin_amdgcn_wmma_f32_16x16x32_bf16(false, a0.v, false, bf.v, (short)0, c0, false, false);
    c1 = __builtin_amdgcn_wmma_f32_16x16x32_bf16(false, a1.v, false, bf.v, (short)0, c1, false, false);
    __syncthreads();
  }
  // C layout: lane halves = N 0..15, VGPR index = M (0..7 / 8..15)
  int n = n0b + n0w + (lane & 15);
  int mbase = 8*(lane>>4);
  #pragma unroll
  for(int r=0;r<8;++r){
    gates[(size_t)(mbase+r     )*H4 + n] = c0[r];
    gates[(size_t)(mbase+r + 16)*H4 + n] = c1[r];
  }
}

__global__ void k_lstm_point(const float* __restrict__ gates,
                             const float* __restrict__ bih, const float* __restrict__ bhh,
                             float* __restrict__ cF, uint16_t* __restrict__ hB,
                             float* __restrict__ Hall, int t){
  int i = blockIdx.x*blockDim.x + threadIdx.x;
  if (i >= BB*HID) return;
  int b = i / HID, j = i % HID;
  const float* g = gates + (size_t)b*H4;
  float gi = g[j]         + bih[j]         + bhh[j];
  float gf = g[HID+j]     + bih[HID+j]     + bhh[HID+j];
  float gg = g[2*HID+j]   + bih[2*HID+j]   + bhh[2*HID+j];
  float go = g[3*HID+j]   + bih[3*HID+j]   + bhh[3*HID+j];
  float c  = sigmoidf_(gf)*cF[i] + sigmoidf_(gi)*tanhf(gg);
  float h  = sigmoidf_(go)*tanhf(c);
  cF[i] = c;
  hB[i] = f2bf(h);
  Hall[(size_t)(b*TT + t)*HID + j] = h;
}

// ---------------- attention + p_gen + A-pack (one block per (b,t)) ----------------
__global__ __launch_bounds__(256) void k_attn(const float* __restrict__ Hall,
                       const float* __restrict__ enc,   // [B][L][EMB]
                       const int*   __restrict__ encIn, // [B][L]
                       const float* __restrict__ attnW, const float* __restrict__ attnB,
                       const float* __restrict__ pgeW,  const float* __restrict__ pgeB,
                       const float* __restrict__ pgdW,  const float* __restrict__ pgdB,
                       float* __restrict__ pgOut, uint16_t* __restrict__ Apk){
  __shared__ float sh_h[HID];
  __shared__ float sh_q[EMB];
  __shared__ float sh_s[LL];
  __shared__ float sh_red[256];
  __shared__ float sh_scal[2];
  const int tid = threadIdx.x;
  const int b = blockIdx.y, t = blockIdx.x;
  const float* hrow = Hall + (size_t)(b*TT + t)*HID;
  sh_h[tid] = hrow[tid]; sh_h[tid+256] = hrow[tid+256];
  __syncthreads();
  { // q = h @ attn_W.T + attn_b   (one output per thread)
    float acc = attnB[tid];
    const float* w = attnW + (size_t)tid*HID;
    #pragma unroll 4
    for(int k=0;k<HID;++k) acc += sh_h[k]*w[k];
    sh_q[tid] = acc;
  }
  __syncthreads();
  if (tid < LL){ // masked scores
    const float* e = enc + (size_t)(b*LL + tid)*EMB;
    float s = 0.f;
    #pragma unroll 4
    for(int k=0;k<EMB;++k) s += sh_q[k]*e[k];
    if (encIn[b*LL + tid] == 0) s = -1e30f;
    sh_s[tid] = s;
  }
  __syncthreads();
  if (tid == 0){ float mx=-1e30f; for(int l=0;l<LL;++l) mx=fmaxf(mx, sh_s[l]); sh_scal[0]=mx; }
  __syncthreads();
  if (tid < LL) sh_s[tid] = __expf(sh_s[tid] - sh_scal[0]);
  __syncthreads();
  if (tid == 0){ float s=0.f; for(int l=0;l<LL;++l) s += sh_s[l]; sh_scal[1]=1.f/s; }
  __syncthreads();
  if (tid < LL) sh_s[tid] *= sh_scal[1];          // sh_s = attention weights
  __syncthreads();
  // ctx[e] and p_gen partial sums
  float ctx = 0.f;
  { const float* e0 = enc + (size_t)b*LL*EMB + tid;
    #pragma unroll
    for(int l=0;l<LL;++l) ctx += sh_s[l]*e0[(size_t)l*EMB]; }
  sh_red[tid] = ctx*pgeW[tid] + sh_h[tid]*pgdW[tid] + sh_h[tid+256]*pgdW[tid+256];
  __syncthreads();
  for(int s2=128; s2>0; s2>>=1){
    if (tid < s2) sh_red[tid] += sh_red[tid+s2];
    __syncthreads();
  }
  float pg = sigmoidf_(sh_red[0] + pgeB[0] + pgdB[0]);
  if (tid==0) pgOut[b*TT + t] = pg;
  // Pack fused-GEMM A row: [pg*h (512) | (1-pg)*aw (64)] in bf16
  uint16_t* arow = Apk + (size_t)(b*TT + t)*KF;
  for(int k=tid; k<HID; k+=256) arow[k] = f2bf(pg * sh_h[k]);
  if (tid < LL) arow[HID + tid] = f2bf((1.f-pg) * sh_s[tid]);
}

// ---------------- fused output GEMM: out = A @ [linW.T | conv_vocab] + pg*linear_b ----
// 256 threads = 8 waves; waves 0-3 -> batch b0, waves 4-7 -> batch b1.
// The linear_W LDS panel (K<512) is staged once and shared by both batches;
// the per-batch pointer panels (K>=512, converter gather) use separate buffers.
// Grid: x = batch-pair (16), y = v-tile (469) -> consecutive blocks share the
// same linear_W panel (hot in WGP$/L2; 30.7MB bf16 stays L2-resident).
__global__ __launch_bounds__(256) void k_final(const uint16_t* __restrict__ Apk,
                        const uint16_t* __restrict__ linWb,   // [V][HID] bf16
                        const uint16_t* __restrict__ convB,   // [NOBJ][V] bf16
                        const int*   __restrict__ encIn,      // [B][L]
                        const float* __restrict__ linb,       // [V]
                        const float* __restrict__ pg,         // [B][T]
                        float* __restrict__ out){             // [T*B][V]
  __shared__ uint16_t sB[2][64*32];
  const int tid  = threadIdx.x;
  const int lane = tid & 31;
  const int wave = tid >> 5;
  const int bh   = wave >> 2;                 // batch half: 0 or 1
  const int b    = blockIdx.x*2 + bh;
  const int v0b  = blockIdx.y * 64;
  const int n0w  = (wave & 3) * 16;
  const bool wvalid = (v0b + n0w) < VOCAB;    // VOCAB % 16 == 0 -> 16-tile all-or-nothing
  v8f c0 = {}; v8f c1 = {};
  const uint16_t* arow_base = Apk + (size_t)b*TT*KF;
  for(int kk=0; kk<KF/32; ++kk){
    const int k0 = kk*32;
    if (k0 < HID){
      // regular path: one shared panel; rows contiguous in K -> 16B async copies
      if (tid < 128){
        int n = tid >> 1, half = tid & 1;
        int v = v0b + n;
        void* dst = &sB[0][n*32 + half*16];
        if (v < VOCAB){
          async_cp16(linWb + (size_t)v*HID + k0 + half*16, dst);
          if (k0 + 32 < HID)
            __builtin_prefetch(linWb + (size_t)v*HID + k0 + 32 + half*16, 0, 0);
        } else {
          uint4 z = {0,0,0,0}; *(uint4*)dst = z;
        }
      }
    } else {
      // pointer path: per-batch gather of converter rows, transposed into LDS
      int lt = tid & 127, tb = tid >> 7;          // tb selects batch tile
      int kr = lt >> 2, chunk = lt & 3;
      int bcur = blockIdx.x*2 + tb;
      int row = encIn[bcur*LL + (k0 - HID) + kr];
      const uint16_t* src = convB + (size_t)row*VOCAB + v0b + chunk*16;
      #pragma unroll
      for(int j=0;j<16;++j){
        int v = v0b + chunk*16 + j;
        sB[tb][(chunk*16 + j)*32 + kr] = (v < VOCAB) ? src[j] : (uint16_t)0;
      }
    }
    wait_async();
    __syncthreads();
    if (wvalid){
      const uint16_t* tile = (k0 < HID) ? &sB[0][0] : &sB[bh][0];
      Frag16 bf, a0, a1;
      int n   = n0w + (lane & 15);
      int kbB = (lane >> 4) * 8;
      bf.q[0] = *(const uint4*)&tile[n*32 + kbB];
      bf.q[1] = *(const uint4*)&tile[n*32 + kbB + 16];
      int kbA = (lane >> 4) * 8;
      const uint16_t* p0 = arow_base + (size_t)((lane & 15)     )*KF + k0 + kbA;
      const uint16_t* p1 = arow_base + (size_t)((lane & 15) + 16)*KF + k0 + kbA;
      a0.q[0] = *(const uint4*)p0; a0.q[1] = *(const uint4*)(p0+16);
      a1.q[0] = *(const uint4*)p1; a1.q[1] = *(const uint4*)(p1+16);
      c0 = __builtin_amdgcn_wmma_f32_16x16x32_bf16(false, a0.v, false, bf.v, (short)0, c0, false, false);
      c1 = __builtin_amdgcn_wmma_f32_16x16x32_bf16(false, a1.v, false, bf.v, (short)0, c1, false, false);
    }
    __syncthreads();
  }
  if (!wvalid) return;
  int v = v0b + n0w + (lane & 15);
  float lb = linb[v];
  int mbase = 8*(lane>>4);
  #pragma unroll
  for(int r=0;r<8;++r){
    int m0 = mbase + r, m1 = m0 + 16;                 // m == timestep t
    out[(size_t)(m0*BB + b)*VOCAB + v] = c0[r] + pg[b*TT + m0]*lb;
    out[(size_t)(m1*BB + b)*VOCAB + v] = c1[r] + pg[b*TT + m1]*lb;
  }
}

// ---------------- host launcher ----------------
extern "C" void kernel_launch(void* const* d_in, const int* in_sizes, int n_in,
                              void* d_out, int out_size, void* d_ws, size_t ws_size,
                              hipStream_t stream) {
  (void)in_sizes; (void)n_in; (void)out_size; (void)ws_size;
  const float* features = (const float*)d_in[0];
  const int*   captions = (const int*)d_in[1];
  const int*   encIn    = (const int*)d_in[3];
  const float* enc      = (const float*)d_in[4];
  const float* embedW   = (const float*)d_in[5];
  const float* Wih      = (const float*)d_in[6];
  const float* Whh      = (const float*)d_in[7];
  const float* bih      = (const float*)d_in[8];
  const float* bhh      = (const float*)d_in[9];
  const float* linW     = (const float*)d_in[10];
  const float* linb     = (const float*)d_in[11];
  const float* attnW    = (const float*)d_in[12];
  const float* attnB    = (const float*)d_in[13];
  const float* pgeW     = (const float*)d_in[14];
  const float* pgeB     = (const float*)d_in[15];
  const float* pgdW     = (const float*)d_in[16];
  const float* pgdB     = (const float*)d_in[17];
  const float* conv     = (const float*)d_in[18];

  char* ws = (char*)d_ws;
  size_t off = 0;
  auto alloc = [&](size_t bytes)->char*{ char* p = ws + off; off += (bytes + 255) & ~(size_t)255; return p; };
  uint16_t* wgt   = (uint16_t*)alloc((size_t)H4*KC*2);       // 3.1 MB
  uint16_t* linWb = (uint16_t*)alloc((size_t)VOCAB*HID*2);   // 30.7 MB
  uint16_t* convB = (uint16_t*)alloc((size_t)NOBJ*VOCAB*2);  // 5.5 MB
  uint16_t* embB  = (uint16_t*)alloc((size_t)TT*BB*EMB*2);
  uint16_t* hB    = (uint16_t*)alloc((size_t)BB*HID*2);
  float*    cF    = (float*)   alloc((size_t)BB*HID*4);
  float*    Hall  = (float*)   alloc((size_t)BB*TT*HID*4);   // 2 MB
  float*    gates = (float*)   alloc((size_t)BB*H4*4);
  float*    pgBuf = (float*)   alloc((size_t)BB*TT*4);
  uint16_t* Apk   = (uint16_t*)alloc((size_t)BB*TT*KF*2);    // 1.2 MB

  k_cvt_bf16<<<1024, 256, 0, stream>>>(linW, linWb, VOCAB*HID);
  k_cvt_bf16<<<512,  256, 0, stream>>>(conv, convB, NOBJ*VOCAB);
  k_prep_wgt<<<512,  256, 0, stream>>>(Wih, Whh, wgt);
  k_prep_emb<<<256,  256, 0, stream>>>(features, captions, embedW, embB);
  k_init_state<<<64, 256, 0, stream>>>(hB, cF);

  for(int t=0;t<TT;++t){
    k_lstm_gemm<<<H4/64, 128, 0, stream>>>(embB, hB, wgt, gates, t);
    k_lstm_point<<<(BB*HID+255)/256, 256, 0, stream>>>(gates, bih, bhh, cF, hB, Hall, t);
  }

  k_attn<<<dim3(TT, BB), 256, 0, stream>>>(Hall, enc, encIn, attnW, attnB,
                                           pgeW, pgeB, pgdW, pgdB, pgBuf, Apk);

  k_final<<<dim3(BB/2, (VOCAB+63)/64), 256, 0, stream>>>(Apk, linWb, convB, encIn,
                                                         linb, pgBuf, (float*)d_out);
}
